// RNNLayer_22093311770692
// MI455X (gfx1250) — matrix operand order repaired and checked by reading, not compiled, not run on hardware
//
#include <hip/hip_runtime.h>
#include <hip/hip_bf16.h>
#include <math.h>
#include <stdint.h>

// ---------------------------------------------------------------------------
// LayerNorm-RNN scan for MI455X (gfx1250, wave32, WMMA).
//   P = x @ Wx^T precomputed (parallel), then 512-step serial scan of
//   h = P_t + s @ Wh^T ; s' = gelu(layernorm(h)).
// Batch rows are independent -> 4 workgroups x 16 rows, no cross-WG sync.
// bf16 WMMA (16x16x32) with fp32 accumulate; LN/GELU in fp32.
// Scan kernel: 16 waves/WG, 4 col-tiles per wave, software-pipelined Wh/A
// fragment loads, async global->LDS staging of P_t (ASYNCcnt) overlapped
// with the matmul, in-wave LayerNorm (1 wave = 1 row, shfl_xor reduction).
// ---------------------------------------------------------------------------

typedef __bf16 bf16_t;
typedef __attribute__((ext_vector_type(16))) __bf16 v16bf;
typedef __attribute__((ext_vector_type(8)))  __bf16 v8bf;
typedef __attribute__((ext_vector_type(8)))  float  v8f;
typedef __attribute__((ext_vector_type(4)))  float  v4f;
typedef __attribute__((ext_vector_type(2)))  float  v2f;

#define BB 64
#define TT 512
#define II 1024
#define OO 1024
#define KC 2048   // II + OO

// ---- Kernel 0: W fp32 [O][I+O] -> Wx bf16 [O][I], Wh bf16 [O][O] ----------
__global__ void prep_weights(const float* __restrict__ W,
                             bf16_t* __restrict__ Wx,
                             bf16_t* __restrict__ Wh) {
    int idx = blockIdx.x * blockDim.x + threadIdx.x;
    if (idx >= OO * KC) return;
    int o = idx / KC;
    int c = idx - o * KC;
    float w = W[idx];
    if (c < II) Wx[o * II + c]        = (bf16_t)w;
    else        Wh[o * OO + (c - II)] = (bf16_t)w;
}

// ---- Kernel 1: P[t][b][o] = sum_i x[b][t][i] * Wx[o][i] -------------------
// Block = 256 threads = 8 waves; each wave computes a 16x64 output tile.
__global__ __launch_bounds__(256) void gemm_xWx(const float* __restrict__ x,
                                                const bf16_t* __restrict__ Wx,
                                                float* __restrict__ P) {
    const int lane  = threadIdx.x & 31;
    const int wid   = threadIdx.x >> 5;                 // 0..7
    const int mbase = blockIdx.x * 16;                  // row tile over B*T
    const int nbase = (blockIdx.y * 8 + wid) * 64;      // 64 cols per wave

    const int mrow  = lane & 15;
    const int ahalf = (lane >> 4) * 8;                  // A: K offset 0 / 8
    const int bhalf = (lane >> 4) * 8;                  // B: K offset 0 / 8
    const long arow = (long)(mbase + mrow) * II;

    v8f acc[4] = {{}, {}, {}, {}};

    for (int k = 0; k < II; k += 32) {
        // A fragment: lane holds row mrow, K = k+ahalf+{0..7} and +16..23
        v4f a0 = *(const v4f*)(x + arow + k + ahalf);
        v4f a1 = *(const v4f*)(x + arow + k + ahalf + 4);
        v4f a2 = *(const v4f*)(x + arow + k + ahalf + 16);
        v4f a3 = *(const v4f*)(x + arow + k + ahalf + 20);
        v16bf a;
#pragma unroll
        for (int e = 0; e < 4; ++e) {
            a[e]      = (bf16_t)a0[e];
            a[e + 4]  = (bf16_t)a1[e];
            a[e + 8]  = (bf16_t)a2[e];
            a[e + 12] = (bf16_t)a3[e];
        }
#pragma unroll
        for (int s = 0; s < 4; ++s) {
            int ocol = nbase + 16 * s + (lane & 15);
            const bf16_t* bp = Wx + (long)ocol * II + k + bhalf;
            v8bf blo = *(const v8bf*)bp;
            v8bf bhi = *(const v8bf*)(bp + 16);
            v16bf bm;
#pragma unroll
            for (int e = 0; e < 8; ++e) { bm[e] = blo[e]; bm[e + 8] = bhi[e]; }
            acc[s] = __builtin_amdgcn_wmma_f32_16x16x32_bf16(
                false, a, false, bm, (short)0, acc[s], false, false);
        }
    }

    // C layout: lane L, vgpr v -> m = v + 8*(L>=16), n = L&15.
    // Row r = b*512 + t  =>  P index ((t*64)+b)*1024 + col.
#pragma unroll
    for (int s = 0; s < 4; ++s) {
        int col = nbase + 16 * s + (lane & 15);
#pragma unroll
        for (int v = 0; v < 8; ++v) {
            int r = mbase + v + ahalf;
            int t = r & (TT - 1);
            int b = r >> 9;
            P[((long)t * BB + b) * OO + col] = acc[s][v];
        }
    }
}

// ---- Kernel 2: serial scan ------------------------------------------------
// 4 WGs x 512 threads (16 waves). WG g owns batch rows 16g..16g+15.
// Matmul phase: wave w computes col tiles 4w..4w+3 (cols 64w..64w+63).
// LN phase:     wave w owns row w; h kept in registers, shfl_xor reduction.
// P_t row w is async-copied global->LDS by wave w, overlapped with matmul.
__global__ __launch_bounds__(512) void recurrent_scan(
    const float* __restrict__ P, const bf16_t* __restrict__ Wh,
    const float* __restrict__ gamma, const float* __restrict__ beta,
    const float* __restrict__ init_state, float* __restrict__ out) {

    __shared__ bf16_t s_state[16 * OO];   // 32 KB  carried state (bf16 operand)
    __shared__ float  s_h[16 * OO];       // 64 KB  this step's pre-LN h
    __shared__ float  s_pt[16 * OO];      // 64 KB  async-staged P_t slab

    const int tid      = threadIdx.x;
    const int lane     = tid & 31;
    const int wid      = tid >> 5;        // 0..15 waves
    const int rowsBase = blockIdx.x * 16; // global batch-row base

    // one-time staging
    for (int i = tid; i < 16 * OO; i += 512)
        s_state[i] = (bf16_t)init_state[i & (OO - 1)];
    __syncthreads();

    const int mrow   = lane & 15;
    const int ahalf  = (lane >> 4) * 8;
    const int bhalf  = (lane >> 4) * 8;
    const bf16_t* srow = s_state + mrow * OO;

    // per-wave B base pointers (col fixed per lane, k advances)
    const bf16_t* bbase[4];
#pragma unroll
    for (int s = 0; s < 4; ++s) {
        int ocol = (wid * 4 + s) * 16 + (lane & 15);
        bbase[s] = Wh + (long)ocol * OO + bhalf;
    }

    // per-lane addresses for the async P-row copy (ISA: LDS addr = addr[31:0])
    const uint32_t pt_lds = (uint32_t)(uintptr_t)(s_pt + wid * OO) + lane * 16;

    for (int t = 0; t < TT; ++t) {
        const float* Pt = P + ((long)t * BB + rowsBase) * OO;

        // ---- async stage P_t row `wid` into LDS (4 KB/wave, ASYNCcnt) -----
        {
            const uint64_t g0 = (uint64_t)(uintptr_t)(Pt + wid * OO) + lane * 16;
#pragma unroll
            for (int it = 0; it < 8; ++it) {
                uint32_t l = pt_lds + it * 512;
                uint64_t g = g0 + it * 512;
                asm volatile("global_load_async_to_lds_b128 %0, %1, off"
                             :: "v"(l), "v"(g) : "memory");
            }
        }

        // ---- h = state @ Wh^T, software-pipelined fragment loads ----------
        v8f acc[4] = {{}, {}, {}, {}};
        v8bf alo = *(const v8bf*)(srow + ahalf);
        v8bf ahi = *(const v8bf*)(srow + ahalf + 16);
        v8bf blo[4], bhi[4];
#pragma unroll
        for (int s = 0; s < 4; ++s) {
            blo[s] = *(const v8bf*)(bbase[s]);
            bhi[s] = *(const v8bf*)(bbase[s] + 16);
        }
#pragma unroll 1
        for (int k = 0; k < OO; k += 32) {
            v16bf a;
#pragma unroll
            for (int e = 0; e < 8; ++e) { a[e] = alo[e]; a[e + 8] = ahi[e]; }
            v16bf bm[4];
#pragma unroll
            for (int s = 0; s < 4; ++s) {
#pragma unroll
                for (int e = 0; e < 8; ++e) {
                    bm[s][e] = blo[s][e];
                    bm[s][e + 8] = bhi[s][e];
                }
            }
            // preload next iteration (wraps to k=0 on last: harmless reload)
            const int kn = (k + 32) & (OO - 1);
            alo = *(const v8bf*)(srow + kn + ahalf);
            ahi = *(const v8bf*)(srow + kn + ahalf + 16);
#pragma unroll
            for (int s = 0; s < 4; ++s) {
                blo[s] = *(const v8bf*)(bbase[s] + kn);
                bhi[s] = *(const v8bf*)(bbase[s] + kn + 16);
            }
#pragma unroll
            for (int s = 0; s < 4; ++s)
                acc[s] = __builtin_amdgcn_wmma_f32_16x16x32_bf16(
                    false, a, false, bm[s], (short)0, acc[s], false, false);
        }
        // spill h tiles to LDS (disjoint columns per wave)
#pragma unroll
        for (int s = 0; s < 4; ++s) {
            int col = (wid * 4 + s) * 16 + (lane & 15);
#pragma unroll
            for (int v = 0; v < 8; ++v)
                s_h[(v + ahalf) * OO + col] = acc[s][v];
        }
        __syncthreads();

        // wave-local: our async P row must have landed (only we consume it)
        asm volatile("s_wait_asynccnt 0x0" ::: "memory");

        // ---- LayerNorm + GELU: wave `wid` owns row `wid` ------------------
        // lane covers column pairs {2*lane, 2*lane+1} + 64*j, j = 0..15
        float h[32];
        float psum = 0.f, psum2 = 0.f;
        const float* hrow = s_h + wid * OO;
        const float* prow = s_pt + wid * OO;
#pragma unroll
        for (int j = 0; j < 16; ++j) {
            int c = 2 * lane + 64 * j;
            v2f hv = *(const v2f*)(hrow + c);
            v2f pv = *(const v2f*)(prow + c);
            float h0 = hv[0] + pv[0];
            float h1 = hv[1] + pv[1];
            h[2 * j] = h0;
            h[2 * j + 1] = h1;
            psum  += h0 + h1;
            psum2 += h0 * h0 + h1 * h1;
        }
#pragma unroll
        for (int off = 16; off > 0; off >>= 1) {
            psum  += __shfl_xor(psum, off, 32);
            psum2 += __shfl_xor(psum2, off, 32);
        }
        const float mu   = psum * (1.0f / OO);
        const float var  = psum2 * (1.0f / OO) - mu * mu;
        const float rstd = rsqrtf(var + 1e-5f);

        float* orow = out + ((long)(rowsBase + wid) * TT + t) * OO;
        bf16_t* strow = s_state + wid * OO;
#pragma unroll
        for (int j = 0; j < 16; ++j) {
            int c = 2 * lane + 64 * j;
            float g0, g1;
            {
                float hn = (h[2 * j] - mu) * rstd * gamma[c] + beta[c];
                g0 = 0.5f * hn * (1.0f + erff(hn * 0.70710678118654752f));
            }
            {
                float hn = (h[2 * j + 1] - mu) * rstd * gamma[c + 1] + beta[c + 1];
                g1 = 0.5f * hn * (1.0f + erff(hn * 0.70710678118654752f));
            }
            v2f ov; ov[0] = g0; ov[1] = g1;
            *(v2f*)(orow + c) = ov;           // coalesced 8B/lane store
            strow[c]     = (bf16_t)g0;        // merges into ds_store_b32
            strow[c + 1] = (bf16_t)g1;
        }
        __syncthreads();  // protect s_state / s_h / s_pt for next step
    }
}

// ---------------------------------------------------------------------------
extern "C" void kernel_launch(void* const* d_in, const int* in_sizes, int n_in,
                              void* d_out, int out_size, void* d_ws, size_t ws_size,
                              hipStream_t stream) {
    const float* x     = (const float*)d_in[0];
    const float* W     = (const float*)d_in[1];
    const float* gamma = (const float*)d_in[2];
    const float* beta  = (const float*)d_in[3];
    const float* inits = (const float*)d_in[4];
    float* out = (float*)d_out;

    // workspace: [Wx bf16 2MB][Wh bf16 2MB][P fp32 128MB]
    char*   ws = (char*)d_ws;
    bf16_t* Wx = (bf16_t*)ws;
    bf16_t* Wh = (bf16_t*)(ws + (size_t)OO * II * sizeof(bf16_t));
    float*  P  = (float*)(ws + (size_t)2 * OO * II * sizeof(bf16_t));

    prep_weights<<<(OO * KC + 255) / 256, 256, 0, stream>>>(W, Wx, Wh);
    gemm_xWx<<<dim3((BB * TT) / 16, OO / 512), 256, 0, stream>>>(x, Wx, P);
    recurrent_scan<<<dim3(BB / 16), 512, 0, stream>>>(P, Wh, gamma, beta, inits, out);
}
